// AdvancedFuzzyAttention_18322330485191
// MI455X (gfx1250) — matrix operand order repaired and checked by reading, not compiled, not run on hardware
//
#include <hip/hip_runtime.h>

// ---------------------------------------------------------------------------
// Problem constants (from reference): B=8192, HID=1024, H=8, D=128.
// out = (LN(value) @ Wv.T) @ Wo.T + bo + LN(query)   (attn == 1.0 exactly,
// since softmax over a singleton axis is 1).  We fuse W = Wo @ Wv and run a
// single 8192x1024x1024 GEMM via bf16 split-precision WMMA (fp32-accurate).
// W is stored pre-swizzled in a contiguous-k WMMA B-fragment layout (L2-resident,
// coalesced b128 loads); the streaming A operand is moved global->LDS by the
// Tensor Data Mover (tensor_load_to_lds) with double buffering.
// ---------------------------------------------------------------------------

#define BB   8192
#define HID  1024
#define NH   8
#define NBLK (HID / 16)   // 64 n-fragments per k-block row

typedef __attribute__((ext_vector_type(16))) __bf16   v16bf;
typedef __attribute__((ext_vector_type(8)))  float    v8f;
typedef __attribute__((ext_vector_type(4)))  unsigned v4u;
typedef __attribute__((ext_vector_type(8)))  int      v8i;
typedef __attribute__((ext_vector_type(4)))  int      v4i;

__device__ __forceinline__ unsigned short f32_to_bf16_rn(float f) {
    unsigned u = __float_as_uint(f);
    u += 0x7FFFu + ((u >> 16) & 1u);   // round-to-nearest-even
    return (unsigned short)(u >> 16);
}
__device__ __forceinline__ float bf16_to_f32(unsigned short h) {
    return __uint_as_float(((unsigned)h) << 16);
}

// ---------------------------------------------------------------------------
// Kernel 1: LayerNorm(query) -> fp32 (residual), LayerNorm(value) -> bf16
// hi/lo split; also writes attn = 1.0.  One block per row.
// ---------------------------------------------------------------------------
__global__ __launch_bounds__(256)
void ln_prep(const float* __restrict__ query, const float* __restrict__ value,
             const float* __restrict__ lnqw, const float* __restrict__ lnqb,
             const float* __restrict__ lnvw, const float* __restrict__ lnvb,
             float* __restrict__ qout,
             unsigned short* __restrict__ vhi, unsigned short* __restrict__ vlo,
             float* __restrict__ attn)
{
    const int b = blockIdx.x;
    const int t = threadIdx.x;

    const float4 qv = reinterpret_cast<const float4*>(query + (size_t)b * HID)[t];
    const float4 vv = reinterpret_cast<const float4*>(value + (size_t)b * HID)[t];

    float qs  = qv.x + qv.y + qv.z + qv.w;
    float qss = qv.x*qv.x + qv.y*qv.y + qv.z*qv.z + qv.w*qv.w;
    float vs  = vv.x + vv.y + vv.z + vv.w;
    float vss = vv.x*vv.x + vv.y*vv.y + vv.z*vv.z + vv.w*vv.w;

    __shared__ float4 red[256];
    red[t] = make_float4(qs, qss, vs, vss);
    __syncthreads();
    for (int s = 128; s > 0; s >>= 1) {
        if (t < s) {
            float4 a = red[t], c = red[t + s];
            red[t] = make_float4(a.x + c.x, a.y + c.y, a.z + c.z, a.w + c.w);
        }
        __syncthreads();
    }
    const float inv = 1.0f / (float)HID;
    const float qmu = red[0].x * inv;
    const float vmu = red[0].z * inv;
    const float qrstd = __frsqrt_rn(fmaxf(red[0].y * inv - qmu * qmu, 0.f) + 1e-5f);
    const float vrstd = __frsqrt_rn(fmaxf(red[0].w * inv - vmu * vmu, 0.f) + 1e-5f);

    const float qe[4] = {qv.x, qv.y, qv.z, qv.w};
    const float ve[4] = {vv.x, vv.y, vv.z, vv.w};
    float4 qo;
    float* qop = &qo.x;
    #pragma unroll
    for (int e = 0; e < 4; ++e) {
        const int i = t * 4 + e;
        qop[e] = (qe[e] - qmu) * qrstd * lnqw[i] + lnqb[i];
        const float vn = (ve[e] - vmu) * vrstd * lnvw[i] + lnvb[i];
        const unsigned short hi = f32_to_bf16_rn(vn);
        vhi[(size_t)b * HID + i] = hi;
        vlo[(size_t)b * HID + i] = f32_to_bf16_rn(vn - bf16_to_f32(hi));
    }
    reinterpret_cast<float4*>(qout + (size_t)b * HID)[t] = qo;

    if (t < NH) attn[(size_t)b * NH + t] = 1.0f;   // softmax over singleton axis
}

// ---------------------------------------------------------------------------
// Kernel 2: W = Wo @ Wv (1024^3, fp32 LDS-tiled).  Result written hi/lo in the
// contiguous-k WMMA 16-bit B-fragment layout (matches load_frag_A): fragment (kb, nb) is 512
// bf16 at offset (kb*NBLK+nb)*512; element (k,n) sits at lane = (n&15) +
// ((k&16)?16:0), e = k&15.  One fragment = 32 lanes x 32B = 1KB contiguous.
// ---------------------------------------------------------------------------
__global__ __launch_bounds__(256)
void fuse_w(const float* __restrict__ Wo, const float* __restrict__ Wv,
            unsigned short* __restrict__ wthi, unsigned short* __restrict__ wtlo)
{
    __shared__ float sA[16][17];
    __shared__ float sB[16][17];
    const int tx = threadIdx.x, ty = threadIdx.y;
    const int j = blockIdx.y * 16 + ty;   // row of Wo / W  -> n
    const int i = blockIdx.x * 16 + tx;   // col of Wv / W  -> k
    float acc = 0.f;
    for (int kk = 0; kk < HID; kk += 16) {
        sA[ty][tx] = Wo[(size_t)j * HID + kk + tx];
        sB[ty][tx] = Wv[(size_t)(kk + ty) * HID + i];
        __syncthreads();
        #pragma unroll
        for (int u = 0; u < 16; ++u) acc = fmaf(sA[ty][u], sB[u][tx], acc);
        __syncthreads();
    }
    const int kb = i >> 5, kk2 = i & 31;
    const int nb = j >> 4, nn  = j & 15;
    const int lane = nn + ((kk2 & 16) ? 16 : 0);
    const int e    = kk2 & 15;
    const size_t off = ((size_t)(kb * NBLK + nb) << 9) + lane * 16 + e;
    const unsigned short hi = f32_to_bf16_rn(acc);
    wthi[off] = hi;
    wtlo[off] = f32_to_bf16_rn(acc - bf16_to_f32(hi));
}

// ---------------------------------------------------------------------------
// Kernel 3: fused bias b' = Wo @ bv + bo.
// ---------------------------------------------------------------------------
__global__ __launch_bounds__(256)
void bias_fuse(const float* __restrict__ Wo, const float* __restrict__ bv,
               const float* __restrict__ bo, float* __restrict__ biasf)
{
    const int j = blockIdx.x * 256 + threadIdx.x;
    float acc = bo[j];
    for (int k = 0; k < HID; ++k) acc = fmaf(Wo[(size_t)j * HID + k], bv[k], acc);
    biasf[j] = acc;
}

// ---------------------------------------------------------------------------
// TDM: DMA a 16-row x 32-col bf16 tile (row stride HID elements) from global
// memory into LDS.  D# per ISA ch.8: group0 = {count=1, lds_addr, global
// tile address (57b), type=2}; group1 = {data_size=2B, tensor dims 32x16,
// tile dims 32x16, dim0 stride = HID}; groups 2/3 zero (2-D tensor).
// Tracked by TENSORcnt (s_wait_tensorcnt).  This toolchain's builtin takes
// 6 args: (v4u g0, v8i g1, v4i g2, v4i g3, v8i extra, i32 cpol).
// ---------------------------------------------------------------------------
__device__ __forceinline__ void tdm_load_16x32(unsigned lds_addr,
                                               const unsigned short* gsrc)
{
    const unsigned long long ga = (unsigned long long)(uintptr_t)gsrc;
    v4u g0;
    g0[0] = 1u;                                              // count=1 (valid)
    g0[1] = lds_addr;                                        // LDS byte addr
    g0[2] = (unsigned)ga;                                    // global_addr lo
    g0[3] = ((unsigned)(ga >> 32) & 0x01FFFFFFu) | (2u << 30); // addr hi, type=2
    v8i g1;
    g1[0] = 0x00010000;        // wg_mask=0, data_size=1 (2 bytes)
    g1[1] = (int)(32u << 16);  // tensor_dim0 = 32
    g1[2] = (int)(16u << 16);  // tensor_dim0 hi = 0, tensor_dim1 = 16
    g1[3] = (int)(32u << 16);  // tensor_dim1 hi = 0, tile_dim0 = 32
    g1[4] = 16;                // tile_dim1 = 16, tile_dim2 = 0
    g1[5] = HID;               // tensor_dim0_stride = 1024 elements
    g1[6] = 0;
    g1[7] = 0;
    const v4i gz4 = {0, 0, 0, 0};
    const v8i gz8 = {0, 0, 0, 0, 0, 0, 0, 0};
    __builtin_amdgcn_tensor_load_to_lds(g0, g1, gz4, gz4, gz8, 0);
}

// ---------------------------------------------------------------------------
// Kernel 4: out[b,j] = sum_i v[b,i] * W[j,i] + bias[j] + q[b,j]
// Block tile 128(M) x 64(N), 8 waves (4x2), each wave 2x2 16x16 tiles.
// K-step = 32; split-precision: AhiBhi + AhiBlo + AloBhi, fp32 accum.
// A moved global->LDS by the TDM (per-wave 16x32 sub-tiles, double-buffered);
// B fragments read directly from the pre-swizzled global buffer (L2).
// ---------------------------------------------------------------------------
__device__ __forceinline__ v16bf load_frag_A(const unsigned short* As, int m0, int lane)
{
    const int m  = m0 + (lane & 15);
    const int kb = (lane < 16) ? 0 : 16;
    union { v16bf v; unsigned u32[8]; } f;
    const unsigned* row = reinterpret_cast<const unsigned*>(As + m * 32);
    #pragma unroll
    for (int r = 0; r < 8; ++r) {
        const int kstart = kb + 2 * r;
        f.u32[r] = row[kstart >> 1];    // pairs fuse into 2x ds_load_b128
    }
    return f.v;
}

__device__ __forceinline__ v16bf load_frag_Bg(const unsigned short* __restrict__ wt,
                                              int kb, int nb, int lane)
{
    const uint4* p = reinterpret_cast<const uint4*>(
        wt + (((size_t)(kb * NBLK + nb)) << 9) + lane * 16);
    union { v16bf v; uint4 u128[2]; } f;
    f.u128[0] = p[0];
    f.u128[1] = p[1];
    return f.v;
}

__global__ __launch_bounds__(256)
void gemm_wmma(const unsigned short* __restrict__ vhi,
               const unsigned short* __restrict__ vlo,
               const unsigned short* __restrict__ wthi,
               const unsigned short* __restrict__ wtlo,
               const float* __restrict__ biasf,
               const float* __restrict__ qres,
               float* __restrict__ out)
{
    __shared__ __align__(16) unsigned short As_hi[2][128 * 32];
    __shared__ __align__(16) unsigned short As_lo[2][128 * 32];

    const int t    = threadIdx.x;
    const int lane = t & 31;
    const int wave = t >> 5;           // 0..7
    const int wm   = wave >> 1;        // 0..3
    const int wn   = wave & 1;         // 0..1
    const int mbase = blockIdx.y * 128;
    const int nbase = blockIdx.x * 64;
    const int nb0   = (nbase >> 4) + wn * 2;   // first 16-wide n-fragment index

    // Per-wave TDM sub-tile: 16 rows of the 128-row block tile.
    // LDS flat addresses: low 32 bits are the wave-relative LDS byte offset
    // (ISA 10.2 LDS aperture mapping).
    const int wrow = wave * 16;
    const unsigned ldsHi[2] = {
        (unsigned)(uintptr_t)&As_hi[0][wrow * 32],
        (unsigned)(uintptr_t)&As_hi[1][wrow * 32] };
    const unsigned ldsLo[2] = {
        (unsigned)(uintptr_t)&As_lo[0][wrow * 32],
        (unsigned)(uintptr_t)&As_lo[1][wrow * 32] };
    const unsigned short* gHi = vhi + (size_t)(mbase + wrow) * HID;
    const unsigned short* gLo = vlo + (size_t)(mbase + wrow) * HID;

    v8f acc[2][2];
    #pragma unroll
    for (int mi = 0; mi < 2; ++mi)
        #pragma unroll
        for (int ni = 0; ni < 2; ++ni)
            #pragma unroll
            for (int r = 0; r < 8; ++r) acc[mi][ni][r] = 0.f;

    // prologue: DMA first K-tile into buffer 0
    tdm_load_16x32(ldsHi[0], gHi);
    tdm_load_16x32(ldsLo[0], gLo);

    for (int k0 = 0; k0 < HID; k0 += 32) {
        const int kb  = k0 >> 5;
        const int buf = kb & 1;

        if (k0 + 32 < HID) {
            // DMA next K-tile into the other buffer, then wait until only
            // those 2 TDM ops remain outstanding (current tile complete).
            tdm_load_16x32(ldsHi[buf ^ 1], gHi + k0 + 32);
            tdm_load_16x32(ldsLo[buf ^ 1], gLo + k0 + 32);
            __builtin_amdgcn_s_wait_tensorcnt(2);
        } else {
            __builtin_amdgcn_s_wait_tensorcnt(0);
        }
        __syncthreads();

        // ---- B fragments straight from pre-swizzled global (L2-resident)
        v16bf bh[2], bl[2];
        #pragma unroll
        for (int ni = 0; ni < 2; ++ni) {
            bh[ni] = load_frag_Bg(wthi, kb, nb0 + ni, lane);
            bl[ni] = load_frag_Bg(wtlo, kb, nb0 + ni, lane);
        }

        v16bf ah[2], al[2];
        #pragma unroll
        for (int mi = 0; mi < 2; ++mi) {
            ah[mi] = load_frag_A(&As_hi[buf][0], wm * 32 + mi * 16, lane);
            al[mi] = load_frag_A(&As_lo[buf][0], wm * 32 + mi * 16, lane);
        }

        #pragma unroll
        for (int mi = 0; mi < 2; ++mi)
            #pragma unroll
            for (int ni = 0; ni < 2; ++ni) {
                acc[mi][ni] = __builtin_amdgcn_wmma_f32_16x16x32_bf16(
                    false, ah[mi], false, bh[ni], (short)0, acc[mi][ni], false, false);
                acc[mi][ni] = __builtin_amdgcn_wmma_f32_16x16x32_bf16(
                    false, ah[mi], false, bl[ni], (short)0, acc[mi][ni], false, false);
                acc[mi][ni] = __builtin_amdgcn_wmma_f32_16x16x32_bf16(
                    false, al[mi], false, bh[ni], (short)0, acc[mi][ni], false, false);
            }
        __syncthreads();   // all waves done reading buf before TDM overwrites it
    }

    // ---- epilogue: + bias[col] + LN(query) residual (C/D layout ISA 7.12.2)
    #pragma unroll
    for (int mi = 0; mi < 2; ++mi) {
        #pragma unroll
        for (int ni = 0; ni < 2; ++ni) {
            const int col = nbase + wn * 32 + ni * 16 + (lane & 15);
            const float bcol = biasf[col];
            #pragma unroll
            for (int r = 0; r < 8; ++r) {
                const int mloc = (lane < 16) ? r : (r + 8);
                const int row  = mbase + wm * 32 + mi * 16 + mloc;
                const size_t g = (size_t)row * HID + col;
                out[g] = acc[mi][ni][r] + bcol + qres[g];
            }
        }
    }
}

// ---------------------------------------------------------------------------
extern "C" void kernel_launch(void* const* d_in, const int* in_sizes, int n_in,
                              void* d_out, int out_size, void* d_ws, size_t ws_size,
                              hipStream_t stream)
{
    (void)in_sizes; (void)n_in; (void)out_size; (void)ws_size;

    const float* query = (const float*)d_in[0];
    const float* value = (const float*)d_in[2];
    const float* Wv    = (const float*)d_in[7];
    const float* bv    = (const float*)d_in[8];
    const float* Wo    = (const float*)d_in[9];
    const float* bo    = (const float*)d_in[10];
    const float* lnqw  = (const float*)d_in[11];
    const float* lnqb  = (const float*)d_in[12];
    const float* lnvw  = (const float*)d_in[15];
    const float* lnvb  = (const float*)d_in[16];

    char* ws = (char*)d_ws;
    float*          q     = (float*)ws;           ws += (size_t)BB * HID * sizeof(float);
    unsigned short* vhi   = (unsigned short*)ws;  ws += (size_t)BB * HID * sizeof(unsigned short);
    unsigned short* vlo   = (unsigned short*)ws;  ws += (size_t)BB * HID * sizeof(unsigned short);
    unsigned short* wthi  = (unsigned short*)ws;  ws += (size_t)HID * HID * sizeof(unsigned short);
    unsigned short* wtlo  = (unsigned short*)ws;  ws += (size_t)HID * HID * sizeof(unsigned short);
    float*          biasf = (float*)ws;

    float* out  = (float*)d_out;
    float* attn = out + (size_t)BB * HID;

    ln_prep<<<BB, 256, 0, stream>>>(query, value, lnqw, lnqb, lnvw, lnvb,
                                    q, vhi, vlo, attn);
    fuse_w<<<dim3(HID / 16, HID / 16), dim3(16, 16), 0, stream>>>(Wo, Wv, wthi, wtlo);
    bias_fuse<<<HID / 256, 256, 0, stream>>>(Wo, bv, bo, biasf);
    gemm_wmma<<<dim3(HID / 64, BB / 128), 256, 0, stream>>>(vhi, vlo, wthi, wtlo,
                                                            biasf, q, out);
}
